// GotenInteractionLayer_27805618275235
// MI455X (gfx1250) — matrix operand order repaired and checked by reading, not compiled, not run on hardware
//
#include <hip/hip_runtime.h>
#include <hip/hip_bf16.h>
#include <math.h>

#define NN 20000
#define NE 160000
#define LAT 128

typedef __attribute__((ext_vector_type(16))) __bf16 v16bf;
typedef __attribute__((ext_vector_type(8)))  float  v8f;

__device__ __forceinline__ unsigned short f2bf(float f) {
    unsigned u = __float_as_uint(f);
    unsigned r = (u + 0x7FFFu + ((u >> 16) & 1u)) >> 16;
    return (unsigned short)r;
}

__device__ __forceinline__ float siluf(float x) { return x / (1.f + __expf(-x)); }

__device__ __forceinline__ unsigned fenc(float x) {
    unsigned u = __float_as_uint(x);
    return (u & 0x80000000u) ? ~u : (u | 0x80000000u);
}
__device__ __forceinline__ float fdec(unsigned u) {
    unsigned v = (u & 0x80000000u) ? (u & 0x7FFFFFFFu) : ~u;
    return __uint_as_float(v);
}

// ---- A fragment (CDNA5 16x16x32 bf16, ISA 7.12.2): lane m(0-15) elems 0..7 =
// K(0..7), 8..15 = K(16..23); lanes 16-31 same M, K offset +8/+24.
// Native __bf16 casts -> hardware packed f32->bf16 converts (RNE).
__device__ __forceinline__ v16bf a_frag(const float* A, int lda, int k0, int lane) {
    int m  = lane & 15;
    int kh = (lane >> 4) << 3;
    const float* r = A + m * lda + k0;
    v16bf a;
#pragma unroll
    for (int j = 0; j < 8; ++j) {
        a[j]     = (__bf16)r[kh + j];
        a[8 + j] = (__bf16)r[16 + kh + j];
    }
    return a;
}

template <int KS>
__device__ __forceinline__ void load_af(v16bf* f, const float* A, int lda, int lane) {
#pragma unroll
    for (int k = 0; k < KS; ++k) f[k] = a_frag(A, lda, k * 32, lane);
}

// ---- B fragments pre-packed fragment-major: for (kb, nt), 32 lanes x 16 bf16
// contiguous per lane -> one 32-byte vector load (2x global_load_b128).
template <int KS>
__device__ __forceinline__ v8f gemm_nt(const v16bf* af, const unsigned short* __restrict__ Wp,
                                       int NT, int nt, int lane) {
    v8f acc = {0.f, 0.f, 0.f, 0.f, 0.f, 0.f, 0.f, 0.f};
#pragma unroll
    for (int k = 0; k < KS; ++k) {
        const v16bf b = *(const v16bf*)(Wp + ((size_t)(k * NT + nt) * 32 + lane) * 16);
        acc = __builtin_amdgcn_wmma_f32_16x16x32_bf16(false, af[k], false, b, (short)0, acc,
                                                      false, false);
    }
    return acc;
}

// ---------------- weight conversion into packed-fragment layout ----------------
__global__ void k_cvt_pack(const float* __restrict__ src, unsigned short* __restrict__ dst,
                           int K, int N, int Kpad) {
    int i = blockIdx.x * blockDim.x + threadIdx.x;
    int tot = Kpad * N;
    if (i >= tot) return;
    int e    = i & 15;
    int lane = (i >> 4) & 31;
    int rest = i >> 9;          // kb*NT + nt
    int NT   = N >> 4;
    int nt   = rest % NT;
    int kb   = rest / NT;
    int k = kb * 32 + ((lane >> 4) << 3) + (e & 7) + ((e >> 3) << 4);
    int n = nt * 16 + (lane & 15);
    dst[i] = (k < K) ? f2bf(src[(size_t)k * N + n]) : (unsigned short)0;
}

// ---------------- node precompute: q, k, gs, v ----------------
__global__ __launch_bounds__(64) void k_nodeprep(
    const float* __restrict__ h,
    const unsigned short* WqB, const float* bq,
    const unsigned short* WkB, const float* bk,
    const unsigned short* gsW1B, const float* gsb1, const unsigned short* gsW2B, const float* gsb2,
    const unsigned short* gvW1B, const float* gvb1, const unsigned short* gvW2B, const float* gvb2,
    float* qn, float* kn, float* gsn, float* vn) {
    __shared__ float b0[32][136];
    __shared__ float b1[32][136];
    int lane = threadIdx.x & 31, wave = threadIdx.x >> 5;
    int rb = (lane >> 4) << 3;
    int rowb = blockIdx.x * 32;
    for (int i = threadIdx.x; i < 32 * 128; i += 64) {
        int r = i >> 7, c = i & 127;
        b0[r][c] = h[(size_t)(rowb + r) * LAT + c];
    }
    __syncthreads();
    const float* A0 = &b0[wave * 16][0];
    const float* A1 = &b1[wave * 16][0];
    v16bf af[4], ah[4];
    load_af<4>(af, A0, 136, lane);
#pragma unroll 1
    for (int nt = 0; nt < 8; ++nt) {
        int col = nt * 16 + (lane & 15);
        v8f aq = gemm_nt<4>(af, WqB, 8, nt, lane);
        v8f ak = gemm_nt<4>(af, WkB, 8, nt, lane);
#pragma unroll
        for (int r = 0; r < 8; ++r) {
            int row = wave * 16 + rb + r;
            qn[(size_t)(rowb + row) * LAT + col] = aq[r] + bq[col];
            kn[(size_t)(rowb + row) * LAT + col] = ak[r] + bk[col];
        }
    }
    // gs MLP
#pragma unroll 1
    for (int nt = 0; nt < 8; ++nt) {
        int col = nt * 16 + (lane & 15);
        v8f a = gemm_nt<4>(af, gsW1B, 8, nt, lane);
#pragma unroll
        for (int r = 0; r < 8; ++r) b1[wave * 16 + rb + r][col] = siluf(a[r] + gsb1[col]);
    }
    __syncthreads();
    load_af<4>(ah, A1, 136, lane);
#pragma unroll 1
    for (int nt = 0; nt < 8; ++nt) {
        int col = nt * 16 + (lane & 15);
        v8f a = gemm_nt<4>(ah, gsW2B, 8, nt, lane);
#pragma unroll
        for (int r = 0; r < 8; ++r)
            gsn[(size_t)(rowb + wave * 16 + rb + r) * LAT + col] = a[r] + gsb2[col];
    }
    __syncthreads();
    // gv MLP (reuse b1)
#pragma unroll 1
    for (int nt = 0; nt < 8; ++nt) {
        int col = nt * 16 + (lane & 15);
        v8f a = gemm_nt<4>(af, gvW1B, 8, nt, lane);
#pragma unroll
        for (int r = 0; r < 8; ++r) b1[wave * 16 + rb + r][col] = siluf(a[r] + gvb1[col]);
    }
    __syncthreads();
    load_af<4>(ah, A1, 136, lane);
#pragma unroll 1
    for (int nt = 0; nt < 8; ++nt) {
        int col = nt * 16 + (lane & 15);
        v8f a = gemm_nt<4>(ah, gvW2B, 8, nt, lane);
#pragma unroll
        for (int r = 0; r < 8; ++r)
            vn[(size_t)(rowb + wave * 16 + rb + r) * LAT + col] = a[r] + gvb2[col];
    }
}

// ---------------- edge geometry: eq_linear + rejection + dots -> w ----------------
__global__ __launch_bounds__(128) void k_geom(
    const float* __restrict__ X, const float* __restrict__ sph,
    const float* __restrict__ Wtq, const float* __restrict__ Wtk,
    const int* __restrict__ ec, const int* __restrict__ en, float* __restrict__ wbuf) {
    __shared__ float xc[8][144], xn[8][144], sp[8][9];
    int tid = threadIdx.x;
    int eb = blockIdx.x * 8;
    for (int i = tid; i < 8 * 144; i += 128) {
        int e = i / 144, k = i % 144;
        xc[e][k] = X[(size_t)ec[eb + e] * 144 + k];
        xn[e][k] = X[(size_t)en[eb + e] * 144 + k];
    }
    for (int i = tid; i < 72; i += 128) sp[i / 9][i % 9] = sph[(size_t)(eb + i / 9) * 9 + i % 9];
    __syncthreads();
    int el = tid >> 4, j = tid & 15;
    int e = eb + el;
    float q[9], kk[9];
    const int ss[3] = {0, 1, 4}, se[3] = {1, 4, 9};
#pragma unroll
    for (int l = 0; l < 3; ++l) {
        for (int d = ss[l]; d < se[l]; ++d) {
            float aq = 0.f, ak = 0.f;
#pragma unroll
            for (int i = 0; i < 16; ++i) {
                aq += Wtq[l * 256 + j * 16 + i] * xc[el][i * 9 + d];
                ak += Wtk[l * 256 + j * 16 + i] * xn[el][i * 9 + d];
            }
            q[d] = aq; kk[d] = ak;
        }
    }
    float dq = 0.f, dk = 0.f;
#pragma unroll
    for (int d = 0; d < 9; ++d) { dq += q[d] * sp[el][d]; dk += kk[d] * sp[el][d]; }
#pragma unroll
    for (int d = 0; d < 9; ++d) { q[d] -= dq * sp[el][d]; kk[d] -= dk * sp[el][d]; }
    const float inv[3] = {1.f, 0.5773502691896258f, 0.4472135954999579f};
#pragma unroll
    for (int l = 0; l < 3; ++l) {
        float s = 0.f;
        for (int d = ss[l]; d < se[l]; ++d) s += q[d] * kk[d];
        wbuf[(size_t)e * 48 + l * 16 + j] = s * inv[l];
    }
}

// ---------------- edge kernel: t update + alpha + segment max ----------------
__global__ __launch_bounds__(64) void k_edge_t(
    const float* __restrict__ t_in, const float* __restrict__ wbuf,
    const float* __restrict__ qn, const float* __restrict__ kn,
    const int* __restrict__ ec, const int* __restrict__ en,
    const unsigned short* gwW1B, const float* gwb1, const unsigned short* gwW2B, const float* gwb2,
    const unsigned short* gtW1B, const float* gtb1, const unsigned short* gtW2B, const float* gtb2,
    const unsigned short* WreB, const float* __restrict__ d_uc,
    float* __restrict__ t_out, float* __restrict__ alpha, unsigned* __restrict__ amaxE) {
    __shared__ float b0[32][136], b1[32][136], b2[32][136];
    __shared__ int sec[32], sen[32];
    int lane = threadIdx.x & 31, wave = threadIdx.x >> 5;
    int rb = (lane >> 4) << 3;
    int eb = blockIdx.x * 32;
    float uc = d_uc[0];
    float co = rsqrtf(uc * uc + 1.f);
    if (threadIdx.x < 32) {
        sec[threadIdx.x] = ec[eb + threadIdx.x];
        sen[threadIdx.x] = en[eb + threadIdx.x];
    }
    // w tile padded to K=64
    for (int i = threadIdx.x; i < 32 * 64; i += 64) {
        int r = i >> 6, c = i & 63;
        b0[r][c] = (c < 48) ? wbuf[(size_t)(eb + r) * 48 + c] : 0.f;
    }
    __syncthreads();
    const float* A0 = &b0[wave * 16][0];
    const float* A1 = &b1[wave * 16][0];
    const float* A2 = &b2[wave * 16][0];
    v16bf af2[2], af[4];
    load_af<2>(af2, A0, 136, lane);
    // hidden_w = silu(w@gwW1 + b1) -> b1
#pragma unroll 1
    for (int nt = 0; nt < 8; ++nt) {
        int col = nt * 16 + (lane & 15);
        v8f a = gemm_nt<2>(af2, gwW1B, 8, nt, lane);
#pragma unroll
        for (int r = 0; r < 8; ++r) b1[wave * 16 + rb + r][col] = siluf(a[r] + gwb1[col]);
    }
    __syncthreads();
    // outw = hidden@gwW2 + b2 -> b2
    load_af<4>(af, A1, 136, lane);
#pragma unroll 1
    for (int nt = 0; nt < 8; ++nt) {
        int col = nt * 16 + (lane & 15);
        v8f a = gemm_nt<4>(af, gwW2B, 8, nt, lane);
#pragma unroll
        for (int r = 0; r < 8; ++r) b2[wave * 16 + rb + r][col] = a[r] + gwb2[col];
    }
    __syncthreads();
    // t tile -> b0
    for (int i = threadIdx.x; i < 32 * 128; i += 64) {
        int r = i >> 7, c = i & 127;
        b0[r][c] = t_in[(size_t)(eb + r) * LAT + c];
    }
    __syncthreads();
    // hidden_t = silu(t@gtW1 + b1) -> b1
    load_af<4>(af, A0, 136, lane);
#pragma unroll 1
    for (int nt = 0; nt < 8; ++nt) {
        int col = nt * 16 + (lane & 15);
        v8f a = gemm_nt<4>(af, gtW1B, 8, nt, lane);
#pragma unroll
        for (int r = 0; r < 8; ++r) b1[wave * 16 + rb + r][col] = siluf(a[r] + gtb1[col]);
    }
    __syncthreads();
    // t_new = co*t + uc*co*(outw * (hidden@gtW2 + b2)) -> b2 and global
    load_af<4>(af, A1, 136, lane);
#pragma unroll 1
    for (int nt = 0; nt < 8; ++nt) {
        int col = nt * 16 + (lane & 15);
        v8f a = gemm_nt<4>(af, gtW2B, 8, nt, lane);
#pragma unroll
        for (int r = 0; r < 8; ++r) {
            int row = wave * 16 + rb + r;
            float dt = b2[row][col] * (a[r] + gtb2[col]);
            float tn = co * b0[row][col] + uc * co * dt;
            b2[row][col] = tn;
            t_out[(size_t)(eb + row) * LAT + col] = tn;
        }
    }
    __syncthreads();
    // p = silu(tnew@Wre) * q * k -> b1
    load_af<4>(af, A2, 136, lane);
#pragma unroll 1
    for (int nt = 0; nt < 8; ++nt) {
        int col = nt * 16 + (lane & 15);
        v8f a = gemm_nt<4>(af, WreB, 8, nt, lane);
#pragma unroll
        for (int r = 0; r < 8; ++r) {
            int row = wave * 16 + rb + r;
            float ta = siluf(a[r]);
            b1[row][col] = ta * qn[(size_t)sec[row] * LAT + col] * kn[(size_t)sen[row] * LAT + col];
        }
    }
    __syncthreads();
    // alpha per (edge,head) + segment max via encoded atomic max
    for (int t = threadIdx.x; t < 128; t += 64) {
        int row = t >> 2, hh = t & 3;
        float s = 0.f;
#pragma unroll
        for (int c = 0; c < 32; ++c) s += b1[row][hh * 32 + c];
        s *= 0.1767766952966369f;  // 1/sqrt(32)
        alpha[(size_t)(eb + row) * 4 + hh] = s;
        atomicMax(&amaxE[(size_t)sec[row] * 4 + hh], fenc(s));
    }
}

// ---------------- exp + denominator ----------------
__global__ void k_exden(float* __restrict__ alpha_ex, const unsigned* __restrict__ amaxE,
                        float* __restrict__ den, const int* __restrict__ ec) {
    int gid = blockIdx.x * blockDim.x + threadIdx.x;
    if (gid >= NE * 4) return;
    int e = gid >> 2, hh = gid & 3;
    float am = fdec(amaxE[(size_t)ec[e] * 4 + hh]);
    float ex = __expf(alpha_ex[gid] - am);
    alpha_ex[gid] = ex;
    atomicAdd(&den[(size_t)ec[e] * 4 + hh], ex);
}

// ---------------- edge kernel: spatial + attn + gates + scatters ----------------
__global__ __launch_bounds__(64) void k_edge_gates(
    const float* __restrict__ t_new, const float* __restrict__ exb, const float* __restrict__ den,
    const float* __restrict__ gsn, const float* __restrict__ vn,
    const int* __restrict__ ec, const int* __restrict__ en,
    const float* __restrict__ X, const float* __restrict__ sph,
    const unsigned short* WrsB, const float* rsb,
    const unsigned short* gspW1B, const float* gspb1,
    const unsigned short* gspW2B, const float* gspb2,
    const unsigned short* mdW1B, const float* mdb1, const unsigned short* mdW2B, const float* mdb2,
    const unsigned short* mtW1B, const float* mtb1, const unsigned short* mtW2B, const float* mtb2,
    float* __restrict__ hacc, float* __restrict__ Xacc) {
    __shared__ float b0[32][136], b1[32][136], b2[32][136];
    __shared__ int sec[32], sen[32];
    int lane = threadIdx.x & 31, wave = threadIdx.x >> 5;
    int rb = (lane >> 4) << 3;
    int eb = blockIdx.x * 32;
    if (threadIdx.x < 32) {
        sec[threadIdx.x] = ec[eb + threadIdx.x];
        sen[threadIdx.x] = en[eb + threadIdx.x];
    }
    const float* A0 = &b0[wave * 16][0];
    const float* A1 = &b1[wave * 16][0];
    const float* A2 = &b2[wave * 16][0];
    for (int i = threadIdx.x; i < 32 * 128; i += 64) {
        int r = i >> 7, c = i & 127;
        b0[r][c] = t_new[(size_t)(eb + r) * LAT + c];
    }
    __syncthreads();
    v16bf af[4];
    // combined = spatial + sea -> b1
    load_af<4>(af, A0, 136, lane);
#pragma unroll 1
    for (int nt = 0; nt < 8; ++nt) {
        int col = nt * 16 + (lane & 15);
        v8f a = gemm_nt<4>(af, WrsB, 8, nt, lane);
        int hh = col >> 5;
#pragma unroll
        for (int r = 0; r < 8; ++r) {
            int row = wave * 16 + rb + r;
            int e = eb + row, c_ = sec[row], nb = sen[row];
            float spat = (a[r] + rsb[col]) * gsn[(size_t)nb * LAT + col];
            float attn = exb[(size_t)e * 4 + hh] / den[(size_t)c_ * 4 + hh];
            b1[row][col] = spat + attn * vn[(size_t)nb * LAT + col];
        }
    }
    __syncthreads();
    // hidden = silu(comb@gspW1 + b1) -> b0
    load_af<4>(af, A1, 136, lane);
#pragma unroll 1
    for (int nt = 0; nt < 8; ++nt) {
        int col = nt * 16 + (lane & 15);
        v8f a = gemm_nt<4>(af, gspW1B, 8, nt, lane);
#pragma unroll
        for (int r = 0; r < 8; ++r) b0[wave * 16 + rb + r][col] = siluf(a[r] + gspb1[col]);
    }
    __syncthreads();
    // gates (384 wide): o_s scatter, o_d -> b1, o_t -> b2
    load_af<4>(af, A0, 136, lane);
#pragma unroll 1
    for (int nt = 0; nt < 24; ++nt) {
        int colg = nt * 16 + (lane & 15);
        v8f a = gemm_nt<4>(af, gspW2B, 24, nt, lane);
#pragma unroll
        for (int r = 0; r < 8; ++r) {
            int row = wave * 16 + rb + r;
            float v = a[r] + gspb2[colg];
            if (nt < 8)       atomicAdd(&hacc[(size_t)sec[row] * LAT + colg], v);
            else if (nt < 16) b1[row][colg - 128] = v;
            else              b2[row][colg - 256] = v;
        }
    }
    __syncthreads();
    // hidden_d = silu(o_d@mdW1 + b1) -> b0
    load_af<4>(af, A1, 136, lane);
#pragma unroll 1
    for (int nt = 0; nt < 8; ++nt) {
        int col = nt * 16 + (lane & 15);
        v8f a = gemm_nt<4>(af, mdW1B, 8, nt, lane);
#pragma unroll
        for (int r = 0; r < 8; ++r) b0[wave * 16 + rb + r][col] = siluf(a[r] + mdb1[col]);
    }
    __syncthreads();
    load_af<4>(af, A0, 136, lane);
    v8f gd = gemm_nt<4>(af, mdW2B, 1, 0, lane);
    __syncthreads();
    // hidden_t = silu(o_t@mtW1 + b1) -> b1
    load_af<4>(af, A2, 136, lane);
#pragma unroll 1
    for (int nt = 0; nt < 8; ++nt) {
        int col = nt * 16 + (lane & 15);
        v8f a = gemm_nt<4>(af, mtW1B, 8, nt, lane);
#pragma unroll
        for (int r = 0; r < 8; ++r) b1[wave * 16 + rb + r][col] = siluf(a[r] + mtb1[col]);
    }
    __syncthreads();
    load_af<4>(af, A1, 136, lane);
    v8f gt = gemm_nt<4>(af, mtW2B, 1, 0, lane);
    // scatter dXij = gd*sph + gtt*Xn into Xacc
    int j = lane & 15;
#pragma unroll
    for (int r = 0; r < 8; ++r) {
        int row = wave * 16 + rb + r;
        int e = eb + row, c_ = sec[row], nb = sen[row];
        float gdv = gd[r] + mdb2[j];
        float gtv = gt[r] + mtb2[j];
#pragma unroll
        for (int i = 0; i < 9; ++i) {
            float dx = gdv * sph[(size_t)e * 9 + i] + gtv * X[(size_t)nb * 144 + j * 9 + i];
            atomicAdd(&Xacc[(size_t)c_ * 144 + j * 9 + i], dx);
        }
    }
}

// ---------------- node finalize: h = residual(h, LN(hacc)) ----------------
__global__ __launch_bounds__(128) void k_hfin(const float* __restrict__ hacc,
                                              const float* __restrict__ h_in,
                                              const float* __restrict__ lng,
                                              const float* __restrict__ lnb,
                                              const float* __restrict__ d_uc,
                                              float* __restrict__ h_new) {
    int lane = threadIdx.x & 31, wave = threadIdx.x >> 5;
    int n = blockIdx.x * 4 + wave;
    float uc = d_uc[0], co = rsqrtf(uc * uc + 1.f);
    float v[4];
    float s = 0.f;
#pragma unroll
    for (int i = 0; i < 4; ++i) { v[i] = hacc[(size_t)n * LAT + lane + i * 32]; s += v[i]; }
#pragma unroll
    for (int o = 16; o; o >>= 1) s += __shfl_xor(s, o);
    float mu = s * (1.f / 128.f);
    float s2 = 0.f;
#pragma unroll
    for (int i = 0; i < 4; ++i) { float d = v[i] - mu; s2 += d * d; }
#pragma unroll
    for (int o = 16; o; o >>= 1) s2 += __shfl_xor(s2, o);
    float inv = rsqrtf(s2 * (1.f / 128.f) + 1e-5f);
#pragma unroll
    for (int i = 0; i < 4; ++i) {
        int c = lane + i * 32;
        float dh = (v[i] - mu) * inv * lng[c] + lnb[c];
        h_new[(size_t)n * LAT + c] = co * h_in[(size_t)n * LAT + c] + uc * co * dh;
    }
}

// ---------------- node finalize: X residual + Xp + nrm ----------------
__global__ __launch_bounds__(160) void k_xfin(const float* __restrict__ Xacc,
                                              const float* __restrict__ X_in,
                                              const float* __restrict__ Wvu,
                                              const float* __restrict__ d_uc,
                                              float* __restrict__ X_new,
                                              float* __restrict__ Xp, float* __restrict__ nrm) {
    __shared__ float xa[144], xn2[144], xp[144], rms[3];
    int tid = threadIdx.x, n = blockIdx.x;
    if (tid < 144) xa[tid] = Xacc[(size_t)n * 144 + tid];
    __syncthreads();
    if (tid == 0) {
        const int ss[3] = {0, 1, 4}, se[3] = {1, 4, 9};
        for (int l = 0; l < 3; ++l) {
            float s = 0.f;
            for (int m = 0; m < 16; ++m)
                for (int d = ss[l]; d < se[l]; ++d) { float v = xa[m * 9 + d]; s += v * v; }
            rms[l] = sqrtf(s * (1.f / 16.f) + 1e-8f);
        }
    }
    __syncthreads();
    float uc = d_uc[0], co = rsqrtf(uc * uc + 1.f);
    if (tid < 144) {
        int d = tid % 9;
        int l = (d < 1) ? 0 : ((d < 4) ? 1 : 2);
        float v = co * X_in[(size_t)n * 144 + tid] + uc * co * (xa[tid] / rms[l]);
        xn2[tid] = v;
        X_new[(size_t)n * 144 + tid] = v;
    }
    __syncthreads();
    if (tid < 144) {
        int j = tid / 9, d = tid % 9;
        float s = 0.f;
#pragma unroll
        for (int m = 0; m < 16; ++m) s += xn2[m * 9 + d] * Wvu[m * 16 + j];
        xp[tid] = s;
        Xp[(size_t)n * 144 + tid] = s;
    }
    __syncthreads();
    if (tid < 16) {
        float s = 0.f;
#pragma unroll
        for (int d = 0; d < 9; ++d) { float v = xp[tid * 9 + d]; s += v * v; }
        nrm[(size_t)n * 16 + tid] = sqrtf(s + 1e-12f);
    }
}

// ---------------- gm MLP + final updates ----------------
__global__ __launch_bounds__(64) void k_gm(
    float* __restrict__ h_io, const float* __restrict__ nrm,
    const unsigned short* gmW1B, const float* gmb1,
    const unsigned short* gmW2B, const float* gmb2,
    const float* __restrict__ Xp, float* __restrict__ X_io) {
    __shared__ float b0[32][168];
    __shared__ float b1[32][136];
    int lane = threadIdx.x & 31, wave = threadIdx.x >> 5;
    int rb = (lane >> 4) << 3;
    int nb = blockIdx.x * 32;
    for (int i = threadIdx.x; i < 32 * 160; i += 64) {
        int r = i / 160, c = i % 160;
        float v;
        if (c < 128)      v = h_io[(size_t)(nb + r) * LAT + c];
        else if (c < 144) v = nrm[(size_t)(nb + r) * 16 + (c - 128)];
        else              v = 0.f;
        b0[r][c] = v;
    }
    __syncthreads();
    const float* A0 = &b0[wave * 16][0];
    const float* A1 = &b1[wave * 16][0];
    v16bf af5[5];
    load_af<5>(af5, A0, 168, lane);
#pragma unroll 1
    for (int nt = 0; nt < 8; ++nt) {
        int col = nt * 16 + (lane & 15);
        v8f a = gemm_nt<5>(af5, gmW1B, 8, nt, lane);
#pragma unroll
        for (int r = 0; r < 8; ++r) b1[wave * 16 + rb + r][col] = siluf(a[r] + gmb1[col]);
    }
    __syncthreads();
    v16bf af[4];
    load_af<4>(af, A1, 136, lane);
#pragma unroll 1
    for (int nt = 0; nt < 9; ++nt) {
        int colg = nt * 16 + (lane & 15);
        v8f a = gemm_nt<4>(af, gmW2B, 9, nt, lane);
#pragma unroll
        for (int r = 0; r < 8; ++r) {
            int row = wave * 16 + rb + r;
            int n = nb + row;
            float g = a[r] + gmb2[colg];
            if (colg < 128) {
                h_io[(size_t)n * LAT + colg] = b0[row][colg] + g;
            } else {
                int j = colg - 128;
#pragma unroll
                for (int d = 0; d < 9; ++d) {
                    size_t idx = (size_t)n * 144 + j * 9 + d;
                    X_io[idx] = X_io[idx] + g * Xp[idx];
                }
            }
        }
    }
}

// =======================================================================
extern "C" void kernel_launch(void* const* d_in, const int* in_sizes, int n_in,
                              void* d_out, int out_size, void* d_ws, size_t ws_size,
                              hipStream_t stream) {
    (void)in_sizes; (void)n_in; (void)out_size; (void)ws_size;
    const float* h    = (const float*)d_in[0];
    const float* X    = (const float*)d_in[1];
    const float* t_ij = (const float*)d_in[2];
    const float* sph  = (const float*)d_in[3];
    const float* uc   = (const float*)d_in[4];
    const float* Wtq  = (const float*)d_in[5];
    const float* Wtk  = (const float*)d_in[6];
    const float* WrsW = (const float*)d_in[7];  const float* Wrsb = (const float*)d_in[8];
    const float* WqW  = (const float*)d_in[9];  const float* Wqb  = (const float*)d_in[10];
    const float* WkW  = (const float*)d_in[11]; const float* Wkb  = (const float*)d_in[12];
    const float* WreW = (const float*)d_in[13];
    const float* lng  = (const float*)d_in[14]; const float* lnb  = (const float*)d_in[15];
    const float* Wvu  = (const float*)d_in[16];
    const float* gsW1 = (const float*)d_in[17]; const float* gsb1 = (const float*)d_in[18];
    const float* gsW2 = (const float*)d_in[19]; const float* gsb2 = (const float*)d_in[20];
    const float* gvW1 = (const float*)d_in[21]; const float* gvb1 = (const float*)d_in[22];
    const float* gvW2 = (const float*)d_in[23]; const float* gvb2 = (const float*)d_in[24];
    const float* gwW1 = (const float*)d_in[25]; const float* gwb1 = (const float*)d_in[26];
    const float* gwW2 = (const float*)d_in[27]; const float* gwb2 = (const float*)d_in[28];
    const float* gtW1 = (const float*)d_in[29]; const float* gtb1 = (const float*)d_in[30];
    const float* gtW2 = (const float*)d_in[31]; const float* gtb2 = (const float*)d_in[32];
    const float* gspW1= (const float*)d_in[33]; const float* gspb1= (const float*)d_in[34];
    const float* gspW2= (const float*)d_in[35]; const float* gspb2= (const float*)d_in[36];
    const float* mdW1 = (const float*)d_in[37]; const float* mdb1 = (const float*)d_in[38];
    const float* mdW2 = (const float*)d_in[39]; const float* mdb2 = (const float*)d_in[40];
    const float* mtW1 = (const float*)d_in[41]; const float* mtb1 = (const float*)d_in[42];
    const float* mtW2 = (const float*)d_in[43]; const float* mtb2 = (const float*)d_in[44];
    const float* gmW1 = (const float*)d_in[45]; const float* gmb1 = (const float*)d_in[46];
    const float* gmW2 = (const float*)d_in[47]; const float* gmb2 = (const float*)d_in[48];
    const int* ec = (const int*)d_in[49];
    const int* en = (const int*)d_in[50];

    float* out   = (float*)d_out;
    float* h_out = out;                        // NN*128
    float* X_out = out + (size_t)NN * 128;     // NN*144
    float* t_out = X_out + (size_t)NN * 144;   // NE*128

    char* ws = (char*)d_ws;
    size_t off = 0;
    auto alloc = [&](size_t bytes) -> void* {
        void* p = ws + off;
        off = (off + bytes + 255) & ~(size_t)255;
        return p;
    };
    typedef unsigned short us;
    us* WqB   = (us*)alloc(128 * 128 * 2);
    us* WkB   = (us*)alloc(128 * 128 * 2);
    us* WrsB  = (us*)alloc(128 * 128 * 2);
    us* WreB  = (us*)alloc(128 * 128 * 2);
    us* gsW1B = (us*)alloc(128 * 128 * 2);
    us* gsW2B = (us*)alloc(128 * 128 * 2);
    us* gvW1B = (us*)alloc(128 * 128 * 2);
    us* gvW2B = (us*)alloc(128 * 128 * 2);
    us* gwW1B = (us*)alloc(64 * 128 * 2);      // K padded 48 -> 64
    us* gwW2B = (us*)alloc(128 * 128 * 2);
    us* gtW1B = (us*)alloc(128 * 128 * 2);
    us* gtW2B = (us*)alloc(128 * 128 * 2);
    us* gspW1B= (us*)alloc(128 * 128 * 2);
    us* gspW2B= (us*)alloc(128 * 384 * 2);
    us* mdW1B = (us*)alloc(128 * 128 * 2);
    us* mdW2B = (us*)alloc(128 * 16 * 2);
    us* mtW1B = (us*)alloc(128 * 128 * 2);
    us* mtW2B = (us*)alloc(128 * 16 * 2);
    us* gmW1B = (us*)alloc(160 * 128 * 2);     // K padded 144 -> 160
    us* gmW2B = (us*)alloc(128 * 144 * 2);
    float* qn    = (float*)alloc((size_t)NN * 128 * 4);
    float* kn    = (float*)alloc((size_t)NN * 128 * 4);
    float* gsn   = (float*)alloc((size_t)NN * 128 * 4);
    float* vn    = (float*)alloc((size_t)NN * 128 * 4);
    float* wbuf  = (float*)alloc((size_t)NE * 48 * 4);
    float* alpha = (float*)alloc((size_t)NE * 4 * 4);
    unsigned* amaxE = (unsigned*)alloc((size_t)NN * 4 * 4);
    float* den   = (float*)alloc((size_t)NN * 4 * 4);
    float* hacc  = (float*)alloc((size_t)NN * 128 * 4);
    float* Xacc  = (float*)alloc((size_t)NN * 144 * 4);
    float* Xp    = (float*)alloc((size_t)NN * 144 * 4);
    float* nrm   = (float*)alloc((size_t)NN * 16 * 4);

    auto cvt = [&](const float* s, us* d, int K, int Nc, int Kpad) {
        int tot = Kpad * Nc;
        k_cvt_pack<<<(tot + 255) / 256, 256, 0, stream>>>(s, d, K, Nc, Kpad);
    };
    cvt(WqW, WqB, 128, 128, 128);   cvt(WkW, WkB, 128, 128, 128);
    cvt(WrsW, WrsB, 128, 128, 128); cvt(WreW, WreB, 128, 128, 128);
    cvt(gsW1, gsW1B, 128, 128, 128); cvt(gsW2, gsW2B, 128, 128, 128);
    cvt(gvW1, gvW1B, 128, 128, 128); cvt(gvW2, gvW2B, 128, 128, 128);
    cvt(gwW1, gwW1B, 48, 128, 64);   cvt(gwW2, gwW2B, 128, 128, 128);
    cvt(gtW1, gtW1B, 128, 128, 128); cvt(gtW2, gtW2B, 128, 128, 128);
    cvt(gspW1, gspW1B, 128, 128, 128); cvt(gspW2, gspW2B, 128, 384, 128);
    cvt(mdW1, mdW1B, 128, 128, 128); cvt(mdW2, mdW2B, 128, 16, 128);
    cvt(mtW1, mtW1B, 128, 128, 128); cvt(mtW2, mtW2B, 128, 16, 128);
    cvt(gmW1, gmW1B, 144, 128, 160); cvt(gmW2, gmW2B, 128, 144, 128);

    hipMemsetAsync(amaxE, 0, (size_t)NN * 4 * 4, stream);
    hipMemsetAsync(den, 0, (size_t)NN * 4 * 4, stream);
    hipMemsetAsync(hacc, 0, (size_t)NN * 128 * 4, stream);
    hipMemsetAsync(Xacc, 0, (size_t)NN * 144 * 4, stream);

    k_nodeprep<<<NN / 32, 64, 0, stream>>>(h, WqB, Wqb, WkB, Wkb,
                                           gsW1B, gsb1, gsW2B, gsb2,
                                           gvW1B, gvb1, gvW2B, gvb2,
                                           qn, kn, gsn, vn);
    k_geom<<<NE / 8, 128, 0, stream>>>(X, sph, Wtq, Wtk, ec, en, wbuf);
    k_edge_t<<<NE / 32, 64, 0, stream>>>(t_ij, wbuf, qn, kn, ec, en,
                                         gwW1B, gwb1, gwW2B, gwb2,
                                         gtW1B, gtb1, gtW2B, gtb2,
                                         WreB, uc, t_out, alpha, amaxE);
    k_exden<<<(NE * 4 + 255) / 256, 256, 0, stream>>>(alpha, amaxE, den, ec);
    k_edge_gates<<<NE / 32, 64, 0, stream>>>(t_out, alpha, den, gsn, vn, ec, en, X, sph,
                                             WrsB, Wrsb, gspW1B, gspb1, gspW2B, gspb2,
                                             mdW1B, mdb1, mdW2B, mdb2,
                                             mtW1B, mtb1, mtW2B, mtb2, hacc, Xacc);
    k_hfin<<<NN / 4, 128, 0, stream>>>(hacc, h, lng, lnb, uc, h_out);
    k_xfin<<<NN, 160, 0, stream>>>(Xacc, X, Wvu, uc, X_out, Xp, nrm);
    k_gm<<<NN / 32, 64, 0, stream>>>(h_out, nrm, gmW1B, gmb1, gmW2B, gmb2, Xp, X_out);
}
